// TinyTransformer_83760452206746
// MI455X (gfx1250) — compile-verified, matrix operand order checked by reading, not run on hardware
//
#include <hip/hip_runtime.h>
#include <hip/hip_bf16.h>

#define HID   1024
#define NH    16
#define HD    64
#define FFND  4096
#define NLAYER 4
#define BB    4
#define TT    1024
#define NTOK  (BB*TT)      // 4096
#define NVOC  32000

typedef __attribute__((ext_vector_type(16))) __bf16 bf16x16;
typedef __attribute__((ext_vector_type(8)))  float  f32x8;
typedef __attribute__((ext_vector_type(4)))  unsigned int u32x4;
typedef __attribute__((ext_vector_type(8)))  int i32x8;
typedef __attribute__((ext_vector_type(4)))  int i32x4;

#if defined(__AMDGCN__) && __has_builtin(__builtin_amdgcn_tensor_load_to_lds) && __has_builtin(__builtin_amdgcn_s_wait_tensorcnt)
#define HAVE_TDM 1
#else
#define HAVE_TDM 0
#endif

union FragAB {
    bf16x16 v;
    unsigned int w[8];
    unsigned short u[16];
};

__device__ __forceinline__ unsigned short f2bf(float f) {
    unsigned int x = __float_as_uint(f);
    unsigned int r = x + 0x7FFFu + ((x >> 16) & 1u);   // round-to-nearest-even
    return (unsigned short)(r >> 16);
}
__device__ __forceinline__ unsigned int pack2bf(float a, float b) {
    return (unsigned int)f2bf(a) | ((unsigned int)f2bf(b) << 16);
}
__device__ __forceinline__ f32x8 zero8() {
    f32x8 z;
#pragma unroll
    for (int i = 0; i < 8; ++i) z[i] = 0.0f;
    return z;
}
// K-offset inside a 16x32 bf16 A-frag / 32x16 B-frag for uint-pair q (0..7)
__device__ __forceinline__ int kpair(int q, int hf) {
    return (q >> 2) * 16 + (q & 3) * 2 + hf * 8;
}
// low 32 bits of a generic pointer to LDS == byte offset in workgroup LDS
__device__ __forceinline__ unsigned lds_addr_of(const void* p) {
    return (unsigned)(unsigned long long)(size_t)p;
}

#if HAVE_TDM
// TDM: DMA a 128-row x 32-element bf16 tile (row stride ldElems) into LDS,
// padding 1 DWORD after every 16 DWORDs -> LDS row stride 34 ushorts.
__device__ __forceinline__ void tdm_tile_load(unsigned ldsOff,
                                              const unsigned short* g,
                                              int ldElems) {
    unsigned long long ga = (unsigned long long)(size_t)g;
    u32x4 g0;
    g0[0] = 1u;                                   // count=1 (valid user D#)
    g0[1] = ldsOff;                               // lds_addr
    g0[2] = (unsigned)ga;                         // global_addr[31:0]
    g0[3] = (unsigned)(ga >> 32) | (2u << 30);    // global_addr[56:32] | type=2
    i32x8 g1;
    g1[0] = (int)((1u << 16)       // data_size = 2 bytes
                | (1u << 20)       // pad_enable
                | (3u << 22));     // pad_interval: every 16 DWORDs (+pad_amount=1 DWORD)
    g1[1] = 0;                                    // abar=0, tensor_dim0[15:0]=0
    g1[2] = (int)0x4000u;                         // tensor_dim0=2^30 (hi16), dim1 lo16=0
    g1[3] = (int)(0x4000u | (32u << 16));         // tensor_dim1=2^30 (hi16), tile_dim0=32
    g1[4] = 128;                                  // tile_dim1=128, tile_dim2=0
    g1[5] = ldElems;                              // tensor_dim0_stride[31:0]
    g1[6] = 0;
    g1[7] = 0;
    i32x4 gz4;
#pragma unroll
    for (int i = 0; i < 4; ++i) gz4[i] = 0;
    i32x8 gz8;
#pragma unroll
    for (int i = 0; i < 8; ++i) gz8[i] = 0;
    // 6-arg form (clang-23 / amdgpu-toolchain lane)
    __builtin_amdgcn_tensor_load_to_lds(g0, g1, gz4, gz4, gz8, 0);
}
#endif

// ---------------------------------------------------------------------------
// Embedding gather: x[row] = emb[ids[row]]  (f32 residual stream)
// ---------------------------------------------------------------------------
__global__ __launch_bounds__(256)
void tt_embed_kernel(const int* __restrict__ ids, const float* __restrict__ emb,
                     float* __restrict__ x) {
    int row = blockIdx.x;
    int id  = ids[row];
    const float4* src = (const float4*)(emb + (size_t)id * HID);
    float4* dst = (float4*)(x + (size_t)row * HID);
    dst[threadIdx.x] = src[threadIdx.x];
}

// ---------------------------------------------------------------------------
// f32 -> bf16 elementwise convert (n multiple of 2048); 8 elems/thread
// ---------------------------------------------------------------------------
__global__ __launch_bounds__(256)
void tt_cvt_bf16_kernel(const float* __restrict__ in, unsigned short* __restrict__ out) {
    size_t i = ((size_t)blockIdx.x * 256 + threadIdx.x) * 8;
    float4 a = *(const float4*)(in + i);
    float4 b = *(const float4*)(in + i + 4);
    uint4 o;
    o.x = pack2bf(a.x, a.y); o.y = pack2bf(a.z, a.w);
    o.z = pack2bf(b.x, b.y); o.w = pack2bf(b.z, b.w);
    *(uint4*)(out + i) = o;
}

// ---------------------------------------------------------------------------
// LayerNorm: one block (256 thr) per row of 1024; f32 in, bf16 out
// ---------------------------------------------------------------------------
__global__ __launch_bounds__(256)
void tt_layernorm_kernel(const float* __restrict__ x, const float* __restrict__ w,
                         const float* __restrict__ b, unsigned short* __restrict__ y) {
    __shared__ float s1[256], s2[256];
    int row = blockIdx.x;
    int t = threadIdx.x;
    float4 vx = ((const float4*)(x + (size_t)row * HID))[t];
    float sum = vx.x + vx.y + vx.z + vx.w;
    float sq  = vx.x*vx.x + vx.y*vx.y + vx.z*vx.z + vx.w*vx.w;
    s1[t] = sum; s2[t] = sq;
    __syncthreads();
    for (int off = 128; off > 0; off >>= 1) {
        if (t < off) { s1[t] += s1[t + off]; s2[t] += s2[t + off]; }
        __syncthreads();
    }
    float mu   = s1[0] * (1.0f / HID);
    float var  = s2[0] * (1.0f / HID) - mu * mu;
    float rstd = rsqrtf(var + 1e-5f);
    float4 wvec = ((const float4*)(w))[t];
    float4 bvec = ((const float4*)(b))[t];
    uint2 o;
    o.x = pack2bf((vx.x - mu) * rstd * wvec.x + bvec.x,
                  (vx.y - mu) * rstd * wvec.y + bvec.y);
    o.y = pack2bf((vx.z - mu) * rstd * wvec.z + bvec.z,
                  (vx.w - mu) * rstd * wvec.w + bvec.w);
    ((uint2*)(y + (size_t)row * HID))[t] = o;
}

// ---------------------------------------------------------------------------
// WMMA GEMM: out[M,N] = epi(A[M,K] @ W[N,K]^T + bias[N]); A/W bf16.
// EPI: 0 = bias, 1 = bias+GELU, 2 = bias+residual.  OBF: 1 = bf16 out.
// Block 256 thr = 8 waves (2 M x 4 N); tile 128x128; K-step 32.
// Tiles staged via Tensor Data Mover (double-buffered) when available.
// ---------------------------------------------------------------------------
template<int EPI, int OBF>
__global__ __launch_bounds__(256)
void tt_gemm_kernel(const unsigned short* __restrict__ A, int lda,
                    const unsigned short* __restrict__ W, int ldw,
                    const float* __restrict__ bias,
                    const float* __restrict__ resid, int ldr,
                    void* __restrict__ outv, int ldo, int K) {
    __shared__ unsigned short As[2][128][34];
    __shared__ unsigned short Ws[2][128][34];

    const int tid  = threadIdx.x;
    const int lane = tid & 31;
    const int wid  = tid >> 5;
    const int wm   = (wid >> 2) * 64;
    const int wn   = (wid & 3) * 32;
    const int hf   = lane >> 4;
    const int r    = lane & 15;

    const int tileM = blockIdx.y * 128;
    const int tileN = blockIdx.x * 128;
    const int nIter = K >> 5;

    f32x8 c[4][2];
#pragma unroll
    for (int i = 0; i < 4; ++i)
#pragma unroll
        for (int j = 0; j < 2; ++j) c[i][j] = zero8();

#if HAVE_TDM
    if (wid == 0) {
        tdm_tile_load(lds_addr_of(&As[0][0][0]), A + (size_t)tileM * lda, lda);
        tdm_tile_load(lds_addr_of(&Ws[0][0][0]), W + (size_t)tileN * ldw, ldw);
    }
#endif

    for (int it = 0; it < nIter; ++it) {
        const int p = it & 1;
#if HAVE_TDM
        if (wid == 0) {
            if (it + 1 < nIter) {
                const int k1 = (it + 1) << 5;
                tdm_tile_load(lds_addr_of(&As[1 - p][0][0]),
                              A + (size_t)tileM * lda + k1, lda);
                tdm_tile_load(lds_addr_of(&Ws[1 - p][0][0]),
                              W + (size_t)tileN * ldw + k1, ldw);
                __builtin_amdgcn_s_wait_tensorcnt(2);   // tile `it` done
            } else {
                __builtin_amdgcn_s_wait_tensorcnt(0);
            }
        }
        __syncthreads();
#else
        // fallback: manual bf16 tile staging into buffer p
        const int k0 = it << 5;
#pragma unroll
        for (int ii = 0; ii < 8; ++ii) {
            int idx = ii * 256 + tid;          // uint index, 2048 per tile
            int row = idx >> 4;
            int j   = idx & 15;
            const unsigned int* pa = (const unsigned int*)(A + (size_t)(tileM + row) * lda + k0);
            const unsigned int* pw = (const unsigned int*)(W + (size_t)(tileN + row) * ldw + k0);
            *(unsigned int*)&As[p][row][2 * j] = pa[j];
            *(unsigned int*)&Ws[p][row][2 * j] = pw[j];
            __builtin_prefetch(pa + 16, 0, 3);
            __builtin_prefetch(pw + 16, 0, 3);
        }
        __syncthreads();
#endif

        FragAB bfr[2];
#pragma unroll
        for (int j = 0; j < 2; ++j) {
            int col = wn + j * 16 + r;
#pragma unroll
            for (int q = 0; q < 8; ++q)
                bfr[j].w[q] = *(const unsigned int*)&Ws[p][col][kpair(q, hf)];
        }
#pragma unroll
        for (int i = 0; i < 4; ++i) {
            FragAB afr;
            int rowi = wm + i * 16 + r;
#pragma unroll
            for (int q = 0; q < 8; ++q)
                afr.w[q] = *(const unsigned int*)&As[p][rowi][kpair(q, hf)];
#pragma unroll
            for (int j = 0; j < 2; ++j)
                c[i][j] = __builtin_amdgcn_wmma_f32_16x16x32_bf16(
                    false, afr.v, false, bfr[j].v, (short)0, c[i][j], false, false);
        }
        __syncthreads();
    }

    // epilogue
#pragma unroll
    for (int i = 0; i < 4; ++i) {
#pragma unroll
        for (int j = 0; j < 2; ++j) {
            int col = tileN + wn + j * 16 + r;
            float bv = bias[col];
#pragma unroll
            for (int v = 0; v < 8; ++v) {
                int row = tileM + wm + i * 16 + v + hf * 8;
                float val = c[i][j][v] + bv;
                if (EPI == 1) val = 0.5f * val * (1.0f + erff(val * 0.70710678118654752f));
                if (EPI == 2) val += resid[(size_t)row * ldr + col];
                if (OBF) ((unsigned short*)outv)[(size_t)row * ldo + col] = f2bf(val);
                else     ((float*)outv)[(size_t)row * ldo + col] = val;
            }
        }
    }
}

// ---------------------------------------------------------------------------
// Flash attention over bf16 qkv [NTOK, 3*HID] (per-token [3][NH][HD]).
// Grid (B*NH, T/64); 128 thr = 4 waves; wave owns 16 q rows. bf16 out.
// ---------------------------------------------------------------------------
__global__ __launch_bounds__(128)
void tt_attention_kernel(const unsigned short* __restrict__ qkv,
                         unsigned short* __restrict__ out) {
    __shared__ unsigned short Ks[32][66];      // [key][d]
    __shared__ unsigned short Vs[64][34];      // [d][key] (transposed)
    __shared__ unsigned short Ps[4][16][34];   // per-wave P scratch

    const int tid  = threadIdx.x;
    const int lane = tid & 31;
    const int wid  = tid >> 5;
    const int hf   = lane >> 4;
    const int r    = lane & 15;

    const int b  = blockIdx.x >> 4;
    const int h  = blockIdx.x & 15;
    const int q0 = blockIdx.y * 64 + wid * 16;
    const int ldq = 3 * HID;
    const float scale = 0.125f;                // 1/sqrt(64)

    // Q fragments: direct bf16 copies (scale applied to S later)
    FragAB qf[2];
#pragma unroll
    for (int kc = 0; kc < 2; ++kc) {
        const unsigned short* qb = qkv + (size_t)(b * TT + q0 + r) * ldq + h * HD + kc * 32;
#pragma unroll
        for (int e = 0; e < 16; ++e) {
            int vv = e >> 1;
            int kk = (vv >> 2) * 16 + (vv & 3) * 2 + (e & 1) + hf * 8;
            qf[kc].u[e] = qb[kk];
        }
    }

    f32x8 o[4];
#pragma unroll
    for (int dc = 0; dc < 4; ++dc) o[dc] = zero8();
    float m[8], l[8];
#pragma unroll
    for (int v = 0; v < 8; ++v) { m[v] = -3.0e38f; l[v] = 0.0f; }

    for (int kt = 0; kt < TT / 32; ++kt) {
        // stage 32 keys of K and V (V transposed), straight bf16 copies
#pragma unroll
        for (int ii = 0; ii < 16; ++ii) {
            int flat = ii * 128 + tid;
            int key  = flat >> 6;
            int d    = flat & 63;
            const unsigned short* base =
                qkv + (size_t)(b * TT + kt * 32 + key) * ldq + h * HD + d;
            Ks[key][d] = base[HID];
            Vs[d][key] = base[2 * HID];
        }
        __syncthreads();

        // S = Q @ K^T (two 16-key fragments), then scale
        f32x8 s[2];
#pragma unroll
        for (int sf = 0; sf < 2; ++sf) {
            s[sf] = zero8();
#pragma unroll
            for (int kc = 0; kc < 2; ++kc) {
                FragAB kb;
                int key = sf * 16 + r;
#pragma unroll
                for (int q = 0; q < 8; ++q)
                    kb.w[q] = *(const unsigned int*)&Ks[key][kc * 32 + kpair(q, hf)];
                s[sf] = __builtin_amdgcn_wmma_f32_16x16x32_bf16(
                    false, qf[kc].v, false, kb.v, (short)0, s[sf], false, false);
            }
#pragma unroll
            for (int v = 0; v < 8; ++v) s[sf][v] *= scale;
        }

        // online softmax (row stats live in 16-lane halves)
        float alpha[8];
#pragma unroll
        for (int v = 0; v < 8; ++v) {
            float mx = fmaxf(s[0][v], s[1][v]);
            mx = fmaxf(mx, __shfl_xor(mx, 1, 32));
            mx = fmaxf(mx, __shfl_xor(mx, 2, 32));
            mx = fmaxf(mx, __shfl_xor(mx, 4, 32));
            mx = fmaxf(mx, __shfl_xor(mx, 8, 32));
            float mn = fmaxf(m[v], mx);
            alpha[v] = __expf(m[v] - mn);
            m[v] = mn;
            float p0 = __expf(s[0][v] - mn);
            float p1 = __expf(s[1][v] - mn);
            s[0][v] = p0; s[1][v] = p1;
            float rs = p0 + p1;
            rs += __shfl_xor(rs, 1, 32);
            rs += __shfl_xor(rs, 2, 32);
            rs += __shfl_xor(rs, 4, 32);
            rs += __shfl_xor(rs, 8, 32);
            l[v] = l[v] * alpha[v] + rs;
        }

        // P -> LDS (wave-private), re-load as A fragment
#pragma unroll
        for (int v = 0; v < 8; ++v) {
            int prow = v + hf * 8;
            Ps[wid][prow][r]      = f2bf(s[0][v]);
            Ps[wid][prow][16 + r] = f2bf(s[1][v]);
        }
        FragAB pa;
#pragma unroll
        for (int q = 0; q < 8; ++q)
            pa.w[q] = *(const unsigned int*)&Ps[wid][r][kpair(q, hf)];

        // O = O*alpha + P @ V
#pragma unroll
        for (int dc = 0; dc < 4; ++dc) {
#pragma unroll
            for (int v = 0; v < 8; ++v) o[dc][v] *= alpha[v];
            FragAB vb;
#pragma unroll
            for (int q = 0; q < 8; ++q)
                vb.w[q] = *(const unsigned int*)&Vs[dc * 16 + r][kpair(q, hf)];
            o[dc] = __builtin_amdgcn_wmma_f32_16x16x32_bf16(
                false, pa.v, false, vb.v, (short)0, o[dc], false, false);
        }
        __syncthreads();
    }

    // normalize and write bf16 [B,T,H,HD]
#pragma unroll
    for (int dc = 0; dc < 4; ++dc) {
#pragma unroll
        for (int v = 0; v < 8; ++v) {
            int row = q0 + v + hf * 8;
            out[(size_t)(b * TT + row) * HID + h * HD + dc * 16 + r] = f2bf(o[dc][v] / l[v]);
        }
    }
}

// ---------------------------------------------------------------------------
// Host-side orchestration
// ---------------------------------------------------------------------------
static inline void cvt(const float* in, unsigned short* out, size_t n, hipStream_t s) {
    tt_cvt_bf16_kernel<<<(unsigned)(n / 2048), 256, 0, s>>>(in, out);
}

extern "C" void kernel_launch(void* const* d_in, const int* in_sizes, int n_in,
                              void* d_out, int out_size, void* d_ws, size_t ws_size,
                              hipStream_t stream) {
    (void)in_sizes; (void)n_in; (void)out_size; (void)ws_size;

    const int*   ids    = (const int*)  d_in[0];
    const float* emb    = (const float*)d_in[1];
    const float* ln1_w  = (const float*)d_in[2];
    const float* ln1_b  = (const float*)d_in[3];
    const float* qkv_w  = (const float*)d_in[4];
    const float* qkv_b  = (const float*)d_in[5];
    const float* proj_w = (const float*)d_in[6];
    const float* proj_b = (const float*)d_in[7];
    const float* ln2_w  = (const float*)d_in[8];
    const float* ln2_b  = (const float*)d_in[9];
    const float* fc1_w  = (const float*)d_in[10];
    const float* fc1_b  = (const float*)d_in[11];
    const float* fc2_w  = (const float*)d_in[12];
    const float* fc2_b  = (const float*)d_in[13];
    const float* head_w = (const float*)d_in[14];
    const float* head_b = (const float*)d_in[15];
    float* out = (float*)d_out;

    char* ws = (char*)d_ws;
    float*          x     = (float*)         (ws);                        // 16 MB f32 [NTOK,HID]
    unsigned short* xnbf  = (unsigned short*)(ws + ((size_t)16 << 20));   //  8 MB bf16 [NTOK,HID]
    unsigned short* qkvbf = (unsigned short*)(ws + ((size_t)24 << 20));   // 24 MB bf16 [NTOK,3HID]
    unsigned short* attbf = (unsigned short*)(ws + ((size_t)48 << 20));   //  8 MB bf16 [NTOK,HID]
    unsigned short* hbf   = (unsigned short*)(ws + ((size_t)56 << 20));   // 32 MB bf16 [NTOK,FFND]
    unsigned short* wbuf  = (unsigned short*)(ws + ((size_t)88 << 20));   // 66 MB bf16 weight stage

    tt_embed_kernel<<<NTOK, 256, 0, stream>>>(ids, emb, x);

    for (int l = 0; l < NLAYER; ++l) {
        tt_layernorm_kernel<<<NTOK, 256, 0, stream>>>(x, ln1_w + l * HID, ln1_b + l * HID, xnbf);

        cvt(qkv_w + (size_t)l * 3 * HID * HID, wbuf, (size_t)3 * HID * HID, stream);
        tt_gemm_kernel<0, 1><<<dim3(3 * HID / 128, NTOK / 128), 256, 0, stream>>>(
            xnbf, HID, wbuf, HID, qkv_b + (size_t)l * 3 * HID, nullptr, 0,
            qkvbf, 3 * HID, HID);

        tt_attention_kernel<<<dim3(BB * NH, TT / 64), 128, 0, stream>>>(qkvbf, attbf);

        cvt(proj_w + (size_t)l * HID * HID, wbuf, (size_t)HID * HID, stream);
        tt_gemm_kernel<2, 0><<<dim3(HID / 128, NTOK / 128), 256, 0, stream>>>(
            attbf, HID, wbuf, HID, proj_b + (size_t)l * HID, x, HID, x, HID, HID);

        tt_layernorm_kernel<<<NTOK, 256, 0, stream>>>(x, ln2_w + l * HID, ln2_b + l * HID, xnbf);

        cvt(fc1_w + (size_t)l * FFND * HID, wbuf, (size_t)FFND * HID, stream);
        tt_gemm_kernel<1, 1><<<dim3(FFND / 128, NTOK / 128), 256, 0, stream>>>(
            xnbf, HID, wbuf, HID, fc1_b + (size_t)l * FFND, nullptr, 0, hbf, FFND, HID);

        cvt(fc2_w + (size_t)l * HID * FFND, wbuf, (size_t)HID * FFND, stream);
        tt_gemm_kernel<2, 0><<<dim3(HID / 128, NTOK / 128), 256, 0, stream>>>(
            hbf, FFND, wbuf, FFND, fc2_b + (size_t)l * HID, x, HID, x, HID, FFND);
    }

    cvt(x, xnbf, (size_t)NTOK * HID, stream);
    cvt(head_w, wbuf, (size_t)NVOC * HID, stream);
    tt_gemm_kernel<0, 0><<<dim3(NVOC / 128, NTOK / 128), 256, 0, stream>>>(
        xnbf, HID, wbuf, HID, head_b, nullptr, 0, out, NVOC, HID);
}
